// CouplingTransform_91319594647687
// MI455X (gfx1250) — matrix operand order repaired and checked by reading, not compiled, not run on hardware
//
#include <hip/hip_runtime.h>
#include <hip/hip_bf16.h>
#include <math.h>

// ---------------------------------------------------------------------------
// Coupling transform, CDNA5 (gfx1250) — bf16 WMMA, M-register-blocked.
//   GEMM1: h = relu(x_even @ W1 + b1)       (32768x512 @ 512x512)
//   GEMM2: params = h @ W2 + b2             (32768x512 @ 512x1024)
//   scale = sigmoid(params[:, :512]+2)+eps; out_odd = x_odd*scale + shift
//   logabsdet = sum(log(scale))
//
// 64-row tile/block, 8 waves; each wave owns ALL 4 M-subtiles and 4 N-tiles
// (B fragment fetched once, reused by 4 WMMAs; W1/W2 stream L2 once/block).
// Weights pre-swizzled fragment-major so a B fragment is one contiguous,
// 32B-aligned block per lane. Fragments are kept in SSA ext-vector form
// (shufflevector + bit_cast, no unions) so LDS reads select ds_load_b128 and
// global reads select global_load_b128 with immediate offsets.
// K-loops partially unrolled (4) to bound the scheduling window (full unroll
// spilled in round 2; unions scalarized to ds_load_2addr_b32 in round 3).
// GEMM2 epilogue: params staged to wave-private LDS, row-remapped pass
// (2 rows/lane) -> float4-dense global traffic, 2-register logdet partials.
// ---------------------------------------------------------------------------

#define F_TOTAL 1024
#define F_HALF  512
#define HID     512
#define MT      64          // rows per block
#define PAD     8           // LDS row padding (halves): stride 1040B -> 4-bank step
#define SAW     (F_HALF + PAD)
#define STW     18          // staging row stride (floats): conflict-free, 8B aligned

typedef __attribute__((ext_vector_type(16))) __bf16         v16bf;
typedef __attribute__((ext_vector_type(8)))  float          v8f;
typedef __attribute__((ext_vector_type(8)))  unsigned short u16x8;
typedef __attribute__((ext_vector_type(16))) unsigned short u16x16;

// A fragment: two 16B chunks at p and p+16 halves (LDS) -> 2x ds_load_b128
__device__ __forceinline__ v16bf load_a_frag(const unsigned short* p) {
  u16x8 c0 = *(const u16x8*)p;
  u16x8 c1 = *(const u16x8*)(p + 16);
  u16x16 w = __builtin_shufflevector(c0, c1, 0, 1, 2, 3, 4, 5, 6, 7,
                                             8, 9, 10, 11, 12, 13, 14, 15);
  return __builtin_bit_cast(v16bf, w);
}

// B fragment: one contiguous 32B block (swizzled weights) -> 2x global_load_b128
__device__ __forceinline__ v16bf load_b_frag(const unsigned short* p) {
  return __builtin_bit_cast(v16bf, *(const u16x16*)p);
}

__device__ __forceinline__ unsigned short f2bf(float f) {
  union { float f; unsigned u; } v; v.f = f;
  unsigned r = v.u + 0x7FFFu + ((v.u >> 16) & 1u);   // round-to-nearest-even
  return (unsigned short)(r >> 16);
}

// --- weight prep: f32 row-major -> bf16, fragment-major lane-contiguous ----
// Element i = ((ntile*16 + ks)*32 + lane)*16 + h  holds
//   W[k][n],  n = ntile*16 + (lane&15),  k = ks*32 + (h>>3)*16 + (lane>>4)*8 + (h&7)
__global__ void prep_weights(const float* __restrict__ W1,
                             const float* __restrict__ W2,
                             unsigned short* __restrict__ W1s,
                             unsigned short* __restrict__ W2s) {
  int i = blockIdx.x * blockDim.x + threadIdx.x;
  if (i < F_HALF * HID) {                       // W1: [k=512][n=512]
    int h    = i & 15;
    int lane = (i >> 4) & 31;
    int f    = i >> 9;
    int ks   = f & 15, nt = f >> 4;
    int n    = nt * 16 + (lane & 15);
    int k    = ks * 32 + (h >> 3) * 16 + (lane >> 4) * 8 + (h & 7);
    W1s[i] = f2bf(W1[(size_t)k * HID + n]);
  }
  if (i < HID * F_TOTAL) {                      // W2: [k=512][n=1024]
    int h    = i & 15;
    int lane = (i >> 4) & 31;
    int f    = i >> 9;
    int ks   = f & 15, nt = f >> 4;
    int n    = nt * 16 + (lane & 15);
    int k    = ks * 32 + (h >> 3) * 16 + (lane >> 4) * 8 + (h & 7);
    W2s[i] = f2bf(W2[(size_t)k * F_TOTAL + n]);
  }
}

__global__ __launch_bounds__(256)
void coupling_kernel(const float* __restrict__ x,
                     const float* __restrict__ b1,
                     const float* __restrict__ b2,
                     const unsigned short* __restrict__ W1s,
                     const unsigned short* __restrict__ W2s,
                     float* __restrict__ out,
                     float* __restrict__ logdet) {
  __shared__ unsigned short sA[MT][SAW];        // identity split, bf16
  __shared__ unsigned short sH[MT][SAW];        // hidden activations, bf16
  __shared__ float sStage[8][2][MT][STW];       // per-wave params staging
  __shared__ float sLog[MT];

  const int tid  = threadIdx.x;
  const int wave = tid >> 5;      // 8 waves (wave32)
  const int lane = tid & 31;
  const int l15  = lane & 15;
  const int hi   = lane >> 4;     // lane half-group (WMMA fragment layout)
  const int row0 = blockIdx.x * MT;

  if (tid < MT) sLog[tid] = 0.0f;

  // ---- load x tile, deinterleave even columns -> bf16 in LDS ----
  {
    const float4* x4 = (const float4*)(x + (size_t)row0 * F_TOTAL);
    for (int idx = tid; idx < MT * (F_TOTAL / 4); idx += 256) {
      int r = idx >> 8;
      int c = idx & 255;
      float4 v = x4[idx];
      sA[r][2 * c]     = f2bf(v.x);
      sA[r][2 * c + 1] = f2bf(v.z);
    }
  }
  __syncthreads();

  // per-wave LDS fragment base pointers: all frag addrs = base + imm
  const unsigned short* const pA = &sA[l15][hi * 8];
  const unsigned short* const pH = &sH[l15][hi * 8];

  // ---- GEMM1: h = relu(identity @ W1 + b1) -> sH (bf16) ----
  for (int j = 0; j < 4; ++j) {
    const int nt = wave * 4 + j;
    const int n0 = nt * 16;
    const float bias = b1[n0 + l15];
    v8f acc[4];
    #pragma unroll
    for (int m = 0; m < 4; ++m)
      #pragma unroll
      for (int r = 0; r < 8; ++r) acc[m][r] = bias;

    const unsigned short* bbase = W1s + ((size_t)nt * 512 + lane) * 16;
    #pragma unroll 4
    for (int ks = 0; ks < 16; ++ks) {
      v16bf b = load_b_frag(bbase + (size_t)ks * 512);
      #pragma unroll
      for (int m = 0; m < 4; ++m) {
        v16bf a = load_a_frag(pA + m * (16 * SAW) + ks * 32);
        acc[m] = __builtin_amdgcn_wmma_f32_16x16x32_bf16(false, a, false, b,
                                                         (short)0, acc[m], false, false);
      }
    }
    #pragma unroll
    for (int m = 0; m < 4; ++m)
      #pragma unroll
      for (int r = 0; r < 8; ++r) {
        float v = acc[m][r] > 0.0f ? acc[m][r] : 0.0f;
        sH[m * 16 + r + hi * 8][n0 + l15] = f2bf(v);
      }
  }
  __syncthreads();

  // ---- GEMM2 (+ staged sigmoid/affine/logdet epilogue) ----
  float* const stS = &sStage[wave][0][0][0];
  float* const stH = &sStage[wave][1][0][0];
  float logrow0 = 0.0f, logrow1 = 0.0f;        // lane's 2 rows, summed over j

  for (int j = 0; j < 4; ++j) {
    const int ntS = wave * 4 + j;              // scale n-tile (cols [0,512))
    const int ntH = ntS + 32;                  // shift n-tile (cols [512,1024))
    const float biasS = b2[ntS * 16 + l15];
    const float biasH = b2[F_HALF + ntS * 16 + l15];
    v8f accS[4], accH[4];
    #pragma unroll
    for (int m = 0; m < 4; ++m)
      #pragma unroll
      for (int r = 0; r < 8; ++r) { accS[m][r] = biasS; accH[m][r] = biasH; }

    const unsigned short* bSb = W2s + ((size_t)ntS * 512 + lane) * 16;
    const unsigned short* bHb = W2s + ((size_t)ntH * 512 + lane) * 16;
    #pragma unroll 4
    for (int ks = 0; ks < 16; ++ks) {
      v16bf bS = load_b_frag(bSb + (size_t)ks * 512);
      v16bf bH = load_b_frag(bHb + (size_t)ks * 512);
      #pragma unroll
      for (int m = 0; m < 4; ++m) {
        v16bf a = load_a_frag(pH + m * (16 * SAW) + ks * 32);
        accS[m] = __builtin_amdgcn_wmma_f32_16x16x32_bf16(false, a, false, bS,
                                                          (short)0, accS[m], false, false);
        accH[m] = __builtin_amdgcn_wmma_f32_16x16x32_bf16(false, a, false, bH,
                                                          (short)0, accH[m], false, false);
      }
    }

    // stage this j's params tile (wave-private LDS; in-order DS pipe)
    #pragma unroll
    for (int m = 0; m < 4; ++m)
      #pragma unroll
      for (int r = 0; r < 8; ++r) {
        const int rr = m * 16 + r + hi * 8;
        stS[rr * STW + l15] = accS[m][r];
        stH[rr * STW + l15] = accH[m][r];
      }

    // row-remapped epilogue: each lane owns 2 full rows x 16 cols
    const int cbase = ntS * 32;                // interleaved column base (2*ntS*16)
    #pragma unroll
    for (int rr = 0; rr < 2; ++rr) {
      const int lrow = lane * 2 + rr;          // 0..63
      const int row  = row0 + lrow;
      const float2* pS = (const float2*)&stS[lrow * STW];
      const float2* pHs = (const float2*)&stH[lrow * STW];
      const float4* px = (const float4*)(x   + (size_t)row * F_TOTAL + cbase);
      float4*       po = (float4*)      (out + (size_t)row * F_TOTAL + cbase);
      float ls = 0.0f;
      #pragma unroll
      for (int q = 0; q < 8; ++q) {
        float2 sv = pS[q];
        float2 hv = pHs[q];
        float4 xv = px[q];
        float s0 = 1.0f / (1.0f + __expf(-(sv.x + 2.0f))) + 0.001f;
        float s1 = 1.0f / (1.0f + __expf(-(sv.y + 2.0f))) + 0.001f;
        float4 ov;
        ov.x = xv.x;                           // identity passthrough (exact)
        ov.y = xv.y * s0 + hv.x;               // transform split
        ov.z = xv.z;
        ov.w = xv.w * s1 + hv.y;
        po[q] = ov;
        ls += __logf(s0) + __logf(s1);
      }
      if (rr == 0) logrow0 += ls; else logrow1 += ls;
    }
  }

  // ---- logabsdet: 2 scalar partials per lane -> LDS atomics -> global ----
  atomicAdd(&sLog[lane * 2],     logrow0);
  atomicAdd(&sLog[lane * 2 + 1], logrow1);
  __syncthreads();
  if (tid < MT) logdet[row0 + tid] = sLog[tid];
}

extern "C" void kernel_launch(void* const* d_in, const int* in_sizes, int n_in,
                              void* d_out, int out_size, void* d_ws, size_t ws_size,
                              hipStream_t stream) {
  const float* x  = (const float*)d_in[0];
  const float* W1 = (const float*)d_in[1];
  const float* b1 = (const float*)d_in[2];
  const float* W2 = (const float*)d_in[3];
  const float* b2 = (const float*)d_in[4];

  float* out = (float*)d_out;
  const int Bsz = in_sizes[0] / F_TOTAL;            // 32768
  float* logdet = out + (size_t)Bsz * F_TOTAL;      // tuple: outputs, logabsdet

  unsigned short* W1s = (unsigned short*)d_ws;      // swizzled bf16 weights
  unsigned short* W2s = W1s + (size_t)F_HALF * HID;

  prep_weights<<<(HID * F_TOTAL + 255) / 256, 256, 0, stream>>>(W1, W2, W1s, W2s);
  coupling_kernel<<<Bsz / MT, 256, 0, stream>>>(x, b1, b2, W1s, W2s, out, logdet);
}
